// Writer_31705448579709
// MI455X (gfx1250) — matrix operand order repaired and checked by reading, not compiled, not run on hardware
//
#include <hip/hip_runtime.h>
#include <math.h>
#include <stdint.h>

typedef __attribute__((ext_vector_type(16))) _Float16 v16h;
typedef __attribute__((ext_vector_type(8)))  float    v8f;

#define B_   32
#define CW_  1024
#define DW_  1024
#define N_   8192
#define W_   64
#define H_   8
#define K_   3
#define HWc_ 128
#define EPS_ 1e-8f

__device__ __forceinline__ float sigmoidf_(float x) { return 1.0f / (1.0f + expf(-x)); }
__device__ __forceinline__ float softplusf_(float x) { return (x > 20.0f) ? x : log1pf(expf(x)); }

// Async copy of 16 bytes global -> LDS (CDNA5 ASYNCcnt path, bypasses VGPRs).
__device__ __forceinline__ void async_b128_to_lds(uint32_t lds_byte_off,
                                                  const void* gaddr)
{
  unsigned long long ga = (unsigned long long)(uintptr_t)gaddr;
  asm volatile("global_load_async_to_lds_b128 %0, %1, off"
               :: "v"(lds_byte_off), "v"(ga) : "memory");
}
__device__ __forceinline__ void wait_async0()
{
  asm volatile("s_wait_asynccnt 0x0" ::: "memory");
}

// ---------------------------------------------------------------------------
// Projection GEMM: out(M=32, N=1024) = X(32, K=1024) @ Wt(K, N) + bias
// One wave per 16x16 output tile, v_wmma_f32_16x16x32_f16, f32->f16 in regs.
// A layout (16-bit 16x32): lane m = L%16, element j -> k = g*16 + hi*8 + 2v + p.
// B layout (16-bit 32x16): lane n = L%16, element j -> k = (j/8)*16 + hi*8 + j%8.
// D layout (f32 16x16):    vgpr r, M = r + 8*hi, N = L%16.
// ---------------------------------------------------------------------------
__global__ __launch_bounds__(32)
void proj_gemm_wmma(const float* __restrict__ X, const float* __restrict__ Wt,
                    const float* __restrict__ bias, float* __restrict__ out,
                    int Kdim, int Ndim)
{
  const int nTiles = Ndim >> 4;
  const int mTile  = blockIdx.x / nTiles;
  const int nTile  = blockIdx.x % nTiles;
  const int lane = threadIdx.x;
  const int mr = lane & 15;
  const int hi = lane >> 4;

  v8f acc = {};
  for (int k0 = 0; k0 < Kdim; k0 += 32) {
    v16h a, bfrag;
#pragma unroll
    for (int j = 0; j < 16; ++j) {
      const int vg = j >> 1, p = j & 1;
      const int k = k0 + ((vg >> 2) << 4) + (hi << 3) + ((vg & 3) << 1) + p;
      a[j] = (_Float16)X[(mTile * 16 + mr) * Kdim + k];
    }
    const int klo = k0 + (hi << 3);
#pragma unroll
    for (int j = 0; j < 16; ++j)
      bfrag[j] = (_Float16)Wt[(klo + ((j >> 3) << 4) + (j & 7)) * Ndim + (nTile * 16 + mr)];
    acc = __builtin_amdgcn_wmma_f32_16x16x32_f16(false, a, false, bfrag,
                                                 (short)0, acc, false, false);
  }
#pragma unroll
  for (int r = 0; r < 8; ++r) {
    const int m = mTile * 16 + (hi << 3) + r;
    const int n = nTile * 16 + mr;
    out[m * Ndim + n] = acc[r] + bias[n];
  }
}

// ---------------------------------------------------------------------------
// Per-head feature projections: key (Wk), erase (We, sigmoid), write_feat (Wf)
// ---------------------------------------------------------------------------
__global__ __launch_bounds__(64)
void head_proj(const float* __restrict__ ctrl, const float* __restrict__ dat,
               const float* __restrict__ Wk, const float* __restrict__ bk,
               const float* __restrict__ We, const float* __restrict__ be,
               const float* __restrict__ Wf, const float* __restrict__ bf,
               float* __restrict__ key, float* __restrict__ erase,
               float* __restrict__ wfeat)
{
  const int b = blockIdx.x >> 3;
  const int h = blockIdx.x & 7;
  const int w = threadIdx.x;
  __shared__ float cbuf[HWc_], dbuf[HWc_];
  for (int c = w; c < HWc_; c += 64) {
    cbuf[c] = ctrl[b * CW_ + h * HWc_ + c];
    dbuf[c] = dat[b * DW_ + h * HWc_ + c];
  }
  __syncthreads();
  float sk = bk[h * W_ + w], se = be[h * W_ + w], sf = bf[h * W_ + w];
  for (int c = 0; c < HWc_; ++c) {
    sk += cbuf[c] * Wk[(h * HWc_ + c) * W_ + w];
    se += cbuf[c] * We[(h * HWc_ + c) * W_ + w];
    sf += dbuf[c] * Wf[(h * HWc_ + c) * W_ + w];
  }
  const int o = (b * H_ + h) * W_ + w;
  key[o] = sk;
  erase[o] = sigmoidf_(se);
  wfeat[o] = sf;
}

// ---------------------------------------------------------------------------
// Scalar gates per (b,h): beta, g, gamma, shift softmax, ||key||.
// ---------------------------------------------------------------------------
__global__ __launch_bounds__(256)
void head_scalars(const float* __restrict__ ctrl, const float* __restrict__ key,
                  const float* __restrict__ Wb, const float* __restrict__ bb,
                  const float* __restrict__ Wg, const float* __restrict__ bg,
                  const float* __restrict__ Ws, const float* __restrict__ bs,
                  const float* __restrict__ Wgam, const float* __restrict__ bgam,
                  float* __restrict__ beta, float* __restrict__ g,
                  float* __restrict__ gamma, float* __restrict__ shift,
                  float* __restrict__ keynorm)
{
  const int t = threadIdx.x;
  if (t >= B_ * H_) return;
  const int b = t / H_, h = t % H_;
  float ab = bb[h], ag = bg[h], agam = bgam[h];
  float as0 = bs[h * K_ + 0], as1 = bs[h * K_ + 1], as2 = bs[h * K_ + 2];
  const float* cr = ctrl + b * CW_ + h * HWc_;
  for (int c = 0; c < HWc_; ++c) {
    const float cv = cr[c];
    ab   += cv * Wb[h * HWc_ + c];
    ag   += cv * Wg[h * HWc_ + c];
    agam += cv * Wgam[h * HWc_ + c];
    as0  += cv * Ws[(h * HWc_ + c) * K_ + 0];
    as1  += cv * Ws[(h * HWc_ + c) * K_ + 1];
    as2  += cv * Ws[(h * HWc_ + c) * K_ + 2];
  }
  beta[t]  = softplusf_(ab);
  g[t]     = sigmoidf_(ag);
  gamma[t] = 1.0f + softplusf_(agam);
  const float mx = fmaxf(as0, fmaxf(as1, as2));
  const float e0 = expf(as0 - mx), e1 = expf(as1 - mx), e2 = expf(as2 - mx);
  const float si = 1.0f / (e0 + e1 + e2);
  shift[t * K_ + 0] = e0 * si;
  shift[t * K_ + 1] = e1 * si;
  shift[t * K_ + 2] = e2 * si;
  float nrm = 0.0f;
  for (int w = 0; w < W_; ++w) { const float kv = key[t * W_ + w]; nrm += kv * kv; }
  keynorm[t] = sqrtf(nrm);
}

// ---------------------------------------------------------------------------
// ||memory[b,n,:]|| for all rows (streams 64MB once; stays hot in 192MB L2).
// ---------------------------------------------------------------------------
__global__ __launch_bounds__(256)
void mem_norms(const float* __restrict__ memory, float* __restrict__ memnorm)
{
  const int row = blockIdx.x * 256 + threadIdx.x;
  const float* mr = memory + (size_t)row * W_;
  float s = 0.0f;
  for (int w = 0; w < W_; ++w) s += mr[w] * mr[w];
  memnorm[row] = sqrtf(s);
}

// ---------------------------------------------------------------------------
// Similarity logits via WMMA, f32 memory rows converted in-register to f16.
// ---------------------------------------------------------------------------
__global__ __launch_bounds__(32)
void sim_wmma(const float* __restrict__ key, const float* __restrict__ memory,
              const float* __restrict__ beta, const float* __restrict__ keynorm,
              const float* __restrict__ memnorm, float* __restrict__ simlog)
{
  const int wave = blockIdx.x;          // B_ * 32 waves
  const int b    = wave >> 5;
  const int wsub = wave & 31;           // 16 nTiles each
  const int lane = threadIdx.x;
  const int mr = lane & 15;
  const int hi = lane >> 4;

  v16h afrag[2];
  const int hclamp = mr & 7;
#pragma unroll
  for (int kk = 0; kk < 2; ++kk) {
#pragma unroll
    for (int j = 0; j < 16; ++j) {
      const int vg = j >> 1, p = j & 1;
      const int k = kk * 32 + ((vg >> 2) << 4) + (hi << 3) + ((vg & 3) << 1) + p;
      const float v = key[b * (H_ * W_) + hclamp * W_ + k];
      afrag[kk][j] = (_Float16)((mr < H_) ? v : 0.0f);
    }
  }

  const float* mb = memory + (size_t)b * N_ * W_;
  for (int t = 0; t < 16; ++t) {
    const int nTile = wsub * 16 + t;
    const int nrow  = nTile * 16 + mr;
    if (t + 1 < 16)
      __builtin_prefetch(mb + (size_t)(nrow + 16) * W_, 0, 3);  // global_prefetch_b8
    v8f acc = {};
#pragma unroll
    for (int kk = 0; kk < 2; ++kk) {
      v16h bfrag;
      const int klo = kk * 32 + (hi << 3);
#pragma unroll
      for (int j = 0; j < 16; ++j)
        bfrag[j] = (_Float16)mb[(size_t)nrow * W_ + klo + ((j >> 3) << 4) + (j & 7)];
      acc = __builtin_amdgcn_wmma_f32_16x16x32_f16(false, afrag[kk], false, bfrag,
                                                   (short)0, acc, false, false);
    }
    if (hi == 0) {            // lanes 0-15 hold M = r (heads 0..7)
#pragma unroll
      for (int r = 0; r < H_; ++r) {
        const int n = nTile * 16 + mr;
        const float denom = keynorm[b * H_ + r] * memnorm[b * N_ + n] + EPS_;
        simlog[((size_t)b * H_ + r) * N_ + n] = beta[b * H_ + r] * (acc[r] / denom);
      }
    }
  }
}

// ---------------------------------------------------------------------------
// Addressing: async-stage the 32KB logit row into LDS (one global pass),
// then softmax, interp, circular conv (K=3), gamma-sharpen + renormalize.
// ---------------------------------------------------------------------------
__global__ __launch_bounds__(256)
void addressing(const float* __restrict__ simlog, const float* __restrict__ prev,
                const float* __restrict__ g, const float* __restrict__ shift,
                const float* __restrict__ gamma, float* __restrict__ outw)
{
  const int bh = blockIdx.x;
  const int t  = threadIdx.x;
  __shared__ float wbuf[N_];
  __shared__ float red[256];
  const float* lg = simlog + (size_t)bh * N_;

  // Async DMA the whole logit row into LDS (2048 x b128, 8 per thread).
  const uint32_t wbase = (uint32_t)(uintptr_t)wbuf;
#pragma unroll
  for (int i = 0; i < N_ / (256 * 4); ++i) {
    const int e4 = i * 256 + t;                    // float4 index
    async_b128_to_lds(wbase + e4 * 16, lg + e4 * 4);
  }
  wait_async0();
  __syncthreads();

  // max over row (from LDS)
  float mx = -INFINITY;
  for (int n = t; n < N_; n += 256) mx = fmaxf(mx, wbuf[n]);
  red[t] = mx; __syncthreads();
  for (int s = 128; s > 0; s >>= 1) { if (t < s) red[t] = fmaxf(red[t], red[t + s]); __syncthreads(); }
  mx = red[0]; __syncthreads();

  // exp in place + sum
  float sm = 0.0f;
  for (int n = t; n < N_; n += 256) { const float e = expf(wbuf[n] - mx); wbuf[n] = e; sm += e; }
  __syncthreads();
  red[t] = sm; __syncthreads();
  for (int s = 128; s > 0; s >>= 1) { if (t < s) red[t] += red[t + s]; __syncthreads(); }
  sm = red[0]; __syncthreads();

  const float gv = g[bh];
  const float* pv = prev + (size_t)bh * N_;
  for (int n = t; n < N_; n += 256)
    wbuf[n] = gv * (wbuf[n] / sm) + (1.0f - gv) * pv[n];
  __syncthreads();

  // roll(w,-1)*s0 + w*s1 + roll(w,+1)*s2  (roll(w,o)[n] = w[(n-o) mod N])
  const float s0 = shift[bh * K_ + 0], s1 = shift[bh * K_ + 1], s2 = shift[bh * K_ + 2];
  const float gam = gamma[bh];
  float pvals[N_ / 256];
  float psum = 0.0f;
  int i = 0;
  for (int n = t; n < N_; n += 256, ++i) {
    const float conv = s0 * wbuf[(n + 1) & (N_ - 1)] + s1 * wbuf[n]
                     + s2 * wbuf[(n - 1) & (N_ - 1)];
    const float p = powf(conv + EPS_, gam);
    pvals[i] = p; psum += p;
  }
  red[t] = psum; __syncthreads();
  for (int s = 128; s > 0; s >>= 1) { if (t < s) red[t] += red[t + s]; __syncthreads(); }
  psum = red[0];
  float* ow = outw + (size_t)bh * N_;
  i = 0;
  for (int n = t; n < N_; n += 256, ++i) ow[n] = pvals[i] / psum;
}

// ---------------------------------------------------------------------------
// Memory update (bandwidth-critical: 64MB in / 64MB out). Per-batch erase and
// write_feat slabs (2KB each) are async-DMA'd into LDS once per block.
//   retained = prod_h ( memory * (1 - erase_h * w_h) ),  added = sum_h wf_h*w_h
// ---------------------------------------------------------------------------
__global__ __launch_bounds__(256)
void memory_update(const float* __restrict__ memory, const float* __restrict__ erase,
                   const float* __restrict__ wfeat, const float* __restrict__ outw,
                   float* __restrict__ outmem)
{
  __shared__ float se[H_ * W_];   // 512 floats
  __shared__ float sf[H_ * W_];
  const int t = threadIdx.x;
  const int idx0 = blockIdx.x * 4;           // first (b,n) row of this block
  const int b = idx0 / N_;                   // all 4 rows share one batch

  // 256 threads -> 128 b128 loads for erase + 128 for wfeat.
  if (t < 128) {
    async_b128_to_lds((uint32_t)(uintptr_t)se + t * 16, erase + b * (H_ * W_) + t * 4);
  } else {
    const int t2 = t - 128;
    async_b128_to_lds((uint32_t)(uintptr_t)sf + t2 * 16, wfeat + b * (H_ * W_) + t2 * 4);
  }
  wait_async0();
  __syncthreads();

  const int idx = idx0 + (t >> 6);           // (b,n) row
  const int w   = t & 63;
  const int n   = idx % N_;
  const float mval = memory[(size_t)idx * W_ + w];
  float ret = 1.0f, add = 0.0f;
#pragma unroll
  for (int h = 0; h < H_; ++h) {
    const float wv = outw[((size_t)b * H_ + h) * N_ + n];
    ret *= mval * (1.0f - se[h * W_ + w] * wv);
    add += sf[h * W_ + w] * wv;
  }
  outmem[(size_t)idx * W_ + w] = ret + add;
}

// ---------------------------------------------------------------------------
extern "C" void kernel_launch(void* const* d_in, const int* in_sizes, int n_in,
                              void* d_out, int out_size, void* d_ws, size_t ws_size,
                              hipStream_t stream)
{
  const float* control = (const float*)d_in[0];
  const float* data    = (const float*)d_in[1];
  const float* memory  = (const float*)d_in[2];
  const float* prevw   = (const float*)d_in[3];
  const float* Wc   = (const float*)d_in[4];  const float* bc   = (const float*)d_in[5];
  const float* Wd   = (const float*)d_in[6];  const float* bd   = (const float*)d_in[7];
  const float* Wk   = (const float*)d_in[8];  const float* bk   = (const float*)d_in[9];
  const float* Wb   = (const float*)d_in[10]; const float* bb   = (const float*)d_in[11];
  const float* Wg   = (const float*)d_in[12]; const float* bg   = (const float*)d_in[13];
  const float* Ws   = (const float*)d_in[14]; const float* bs   = (const float*)d_in[15];
  const float* Wgam = (const float*)d_in[16]; const float* bgam = (const float*)d_in[17];
  const float* We   = (const float*)d_in[18]; const float* be   = (const float*)d_in[19];
  const float* Wf   = (const float*)d_in[20]; const float* bf   = (const float*)d_in[21];

  float* ws     = (float*)d_ws;
  float* ctrl   = ws;                         // 32768
  float* dat    = ctrl   + B_ * CW_;          // 32768
  float* keyb   = dat    + B_ * DW_;          // 16384
  float* eraseb = keyb   + B_ * H_ * W_;      // 16384
  float* wfeatb = eraseb + B_ * H_ * W_;      // 16384
  float* betab  = wfeatb + B_ * H_ * W_;      // 256
  float* gb     = betab  + B_ * H_;           // 256
  float* gammab = gb     + B_ * H_;           // 256
  float* shiftb = gammab + B_ * H_;           // 768
  float* keynb  = shiftb + B_ * H_ * K_;      // 256
  float* memnb  = keynb  + B_ * H_;           // 262144
  float* simb   = memnb  + B_ * N_;           // 2097152  (~10MB total)

  float* out_mem = (float*)d_out;
  float* out_w   = out_mem + (size_t)B_ * N_ * W_;

  proj_gemm_wmma<<<(B_ / 16) * (CW_ / 16), 32, 0, stream>>>(control, Wc, bc, ctrl, CW_, H_ * HWc_);
  proj_gemm_wmma<<<(B_ / 16) * (DW_ / 16), 32, 0, stream>>>(data,    Wd, bd, dat,  DW_, H_ * HWc_);
  head_proj<<<B_ * H_, 64, 0, stream>>>(ctrl, dat, Wk, bk, We, be, Wf, bf,
                                        keyb, eraseb, wfeatb);
  head_scalars<<<1, 256, 0, stream>>>(ctrl, keyb, Wb, bb, Wg, bg, Ws, bs, Wgam, bgam,
                                      betab, gb, gammab, shiftb, keynb);
  mem_norms<<<(B_ * N_) / 256, 256, 0, stream>>>(memory, memnb);
  sim_wmma<<<B_ * 32, 32, 0, stream>>>(keyb, memory, betab, keynb, memnb, simb);
  addressing<<<B_ * H_, 256, 0, stream>>>(simb, prevw, gb, shiftb, gammab, out_w);
  memory_update<<<(B_ * N_) / 4, 256, 0, stream>>>(memory, eraseb, wfeatb, out_w, out_mem);
}
